// SelfAttention_Pseudoformer_86844238725646
// MI455X (gfx1250) — matrix operand order repaired and checked by reading, not compiled, not run on hardware
//
#include <hip/hip_runtime.h>
#include <hip/hip_bf16.h>

// ---------------------------------------------------------------------------
// Pseudoformer self-attention for MI455X (gfx1250, wave32, WMMA).
//
// All GEMMs run through one generic bf16-WMMA kernel (v_wmma_f32_16x16x32_bf16,
// f32 accumulate). f32 inputs are converted to bf16 during vectorized
// (b128) LDS tile staging (v_cvt_pk_bf16_f32 when available). The
// O(B*H*L*L*P) relative-position einsum is replaced by a 15-band prefix-sum
// decomposition (relTab has only 15 distinct values, monotone in distance),
// making the problem ~bandwidth bound on 23.3 TB/s HBM.
// Wave tile: 32x64 -> 8 v_wmma per wave per K-step (2 A-frags x 4 B-frags).
// Staging is split into a load phase (16 b128 loads in flight) and a
// convert/ds_store phase so global loads pipeline instead of serializing.
// ---------------------------------------------------------------------------

typedef __attribute__((ext_vector_type(16))) __bf16 v16bf;
typedef __attribute__((ext_vector_type(2)))  __bf16 v2bf;
typedef __attribute__((ext_vector_type(8)))  float  v8f;

constexpr int B_ = 8, L_ = 2048, C_ = 512, P_ = 64, H_ = 8, DK_ = 64, HD_ = 512, R_ = 15;

// band r -> distance range [dlo, dhi] such that relTab value == r
__constant__ int c_dlo[R_] = {0,1,2,3,4,5,6,7, 9,11,15,23,39, 71,135};
__constant__ int c_dhi[R_] = {0,1,2,3,4,5,6,8,10,14,22,38,70,134,2047};

__device__ __forceinline__ unsigned short f2bf(float f) {
  unsigned int u = __float_as_uint(f);
  u += 0x7FFFu + ((u >> 16) & 1u);   // round-to-nearest-even
  return (unsigned short)(u >> 16);
}

// pack two f32 -> two bf16 in one dword (v_cvt_pk_bf16_f32 when available)
__device__ __forceinline__ unsigned int pk2bf(float x, float y) {
#if __has_builtin(__builtin_amdgcn_cvt_pk_bf16_f32)
  union { v2bf v; unsigned int u; } cv;
  cv.v = __builtin_amdgcn_cvt_pk_bf16_f32(x, y);
  return cv.u;
#else
  return (unsigned int)f2bf(x) | ((unsigned int)f2bf(y) << 16);
#endif
}

#define TM 64
#define TN 64
#define TK 32
#define LDSS 36   // padded row stride (shorts): 72 B rows -> uint2 stores stay 8B-aligned

// C[m,n] = scale * sum_k A[m,k]*B[k,n] + bias[n]
// Per-z offsets: base += (z/zdiv)*s1 + (z%zdiv)*s2  for A, B, C.
// transB: B element [k][n] is read from Bg[n*ldb + k].
// Block: 64 threads = 2 waves; each wave computes a 32x64 strip (2x4 frags).
__global__ __launch_bounds__(64)
void gemm_bf16_wmma(const float* __restrict__ A, long long lda, long long sA1, long long sA2,
                    const float* __restrict__ Bg, long long ldb, long long sB1, long long sB2,
                    int transB,
                    float* __restrict__ Cp, long long ldc, long long sC1, long long sC2,
                    const float* __restrict__ bias, float scale,
                    int M, int N, int K, int zdiv)
{
  __shared__ unsigned short As[TM][LDSS];
  __shared__ unsigned short Bs[TN][LDSS];

  const int z = blockIdx.z;
  A  += (long long)(z / zdiv) * sA1 + (long long)(z % zdiv) * sA2;
  Bg += (long long)(z / zdiv) * sB1 + (long long)(z % zdiv) * sB2;
  Cp += (long long)(z / zdiv) * sC1 + (long long)(z % zdiv) * sC2;

  const int m0 = blockIdx.y * TM;
  const int n0 = blockIdx.x * TN;
  const int tid  = threadIdx.x;
  const int lane = tid & 31;
  const int wave = tid >> 5;
  const int wm   = wave * 32;           // wave's 32-row strip
  const int half = lane >> 4;           // 0: lanes 0-15, 1: lanes 16-31
  const int r    = lane & 15;

  v8f acc[2][4];
  #pragma unroll
  for (int mi = 0; mi < 2; ++mi)
    #pragma unroll
    for (int ni = 0; ni < 4; ++ni)
      #pragma unroll
      for (int e = 0; e < 8; ++e) acc[mi][ni][e] = 0.0f;

  for (int k0 = 0; k0 < K; k0 += TK) {
    // ---- load phase: 16 independent b128 loads, kept in flight ----
    float4 fa[8], fb[8];
    #pragma unroll
    for (int pass = 0; pass < 8; ++pass) {
      int idx = pass * 256 + tid * 4;            // 64 thr * 4 f32 * 8 passes = 2048
      int m = idx >> 5, k = idx & 31;
      fa[pass] = *(const float4*)&A[(long long)(m0 + m) * lda + (k0 + k)];
    }
    if (transB) {                                 // memory is [N,K]
      #pragma unroll
      for (int pass = 0; pass < 8; ++pass) {
        int idx = pass * 256 + tid * 4;
        int n = idx >> 5, k = idx & 31;
        fb[pass] = *(const float4*)&Bg[(long long)(n0 + n) * ldb + (k0 + k)];
      }
    } else {                                      // memory is [K,N]
      #pragma unroll
      for (int pass = 0; pass < 8; ++pass) {
        int idx = pass * 256 + tid * 4;
        int k = idx >> 6, n = idx & 63;
        fb[pass] = *(const float4*)&Bg[(long long)(k0 + k) * ldb + (n0 + n)];
      }
    }

    // ---- convert + LDS store phase ----
    #pragma unroll
    for (int pass = 0; pass < 8; ++pass) {
      int idx = pass * 256 + tid * 4;
      int m = idx >> 5, k = idx & 31;
      *(uint2*)&As[m][k] = make_uint2(pk2bf(fa[pass].x, fa[pass].y),
                                      pk2bf(fa[pass].z, fa[pass].w));
    }
    if (transB) {
      #pragma unroll
      for (int pass = 0; pass < 8; ++pass) {
        int idx = pass * 256 + tid * 4;
        int n = idx >> 5, k = idx & 31;
        *(uint2*)&Bs[n][k] = make_uint2(pk2bf(fb[pass].x, fb[pass].y),
                                        pk2bf(fb[pass].z, fb[pass].w));
      }
    } else {                                      // transpose-scatter into Bs[n][k]
      #pragma unroll
      for (int pass = 0; pass < 8; ++pass) {
        int idx = pass * 256 + tid * 4;
        int k = idx >> 6, n = idx & 63;
        Bs[n + 0][k] = f2bf(fb[pass].x);
        Bs[n + 1][k] = f2bf(fb[pass].y);
        Bs[n + 2][k] = f2bf(fb[pass].z);
        Bs[n + 3][k] = f2bf(fb[pass].w);
      }
    }
    if (k0 + TK < K)  // prefetch next A tile rows (global_prefetch_b8)
      __builtin_prefetch(&A[(long long)(m0 + tid) * lda + (k0 + TK)], 0, 1);
    __syncthreads();

    // ---- gather fragments per ISA 16-bit A/B VGPR layouts ----
    union Frag { v16bf v; unsigned int u[8]; } af[2], bfr[4];
    #pragma unroll
    for (int mi = 0; mi < 2; ++mi) {
      int m = wm + mi * 16 + r;
      #pragma unroll
      for (int v = 0; v < 8; ++v) {
        int kb = ((v < 4) ? 0 : 16) + ((v & 3) << 1) + (half << 3); // A: VGPR v pairs
        af[mi].u[v] = *(const unsigned int*)&As[m][kb];
      }
    }
    #pragma unroll
    for (int ni = 0; ni < 4; ++ni) {
      int n = ni * 16 + r;
      #pragma unroll
      for (int v = 0; v < 8; ++v) {
        int kb = ((v < 4) ? 0 : 16) + ((v & 3) << 1) + (half << 3); // B: K pairs
        bfr[ni].u[v] = *(const unsigned int*)&Bs[n][kb];
      }
    }
    #pragma unroll
    for (int mi = 0; mi < 2; ++mi)
      #pragma unroll
      for (int ni = 0; ni < 4; ++ni)
        acc[mi][ni] = __builtin_amdgcn_wmma_f32_16x16x32_bf16(
            false, af[mi].v, false, bfr[ni].v, (short)0, acc[mi][ni], false, false);
    __syncthreads();
  }

  // ---- epilogue: C/D layout lane r = N, vgpr v + 8*half = M ----
  #pragma unroll
  for (int mi = 0; mi < 2; ++mi)
    #pragma unroll
    for (int ni = 0; ni < 4; ++ni) {
      int gn = n0 + ni * 16 + r;
      float bv = bias ? bias[gn] : 0.0f;
      #pragma unroll
      for (int v = 0; v < 8; ++v) {
        int gm = m0 + wm + mi * 16 + half * 8 + v;
        Cp[(long long)gm * ldc + gn] = acc[mi][ni][v] * scale + bv;
      }
    }
}

// Mask + softmax over L (axis=1) per (b,p); also writes masked pScore (output)
// and pAlpha transposed [B,P,L] (f32) for cumsum / kxc GEMM.
__global__ __launch_bounds__(256)
void mask_softmax_L(float* __restrict__ pscore,        // [B,L,P], in/out
                    const int* __restrict__ maskPAD,   // [B,L,L]
                    float* __restrict__ pAlphaT)       // [B,P,L]
{
  const int b = blockIdx.x / P_;
  const int p = blockIdx.x % P_;
  const int t = threadIdx.x;
  __shared__ float red[256];

  float vals[8];
  float mx = -3.4e38f;
  #pragma unroll
  for (int i = 0; i < 8; ++i) {
    int l = t + i * 256;
    int mval = maskPAD[(size_t)b * L_ * L_ + l];     // maskPAD[b,0,l]
    float x = pscore[((size_t)b * L_ + l) * P_ + p];
    x = (mval == 0) ? -32768.0f : x;
    pscore[((size_t)b * L_ + l) * P_ + p] = x;       // reference returns masked pScore
    vals[i] = x;
    mx = fmaxf(mx, x);
  }
  red[t] = mx; __syncthreads();
  for (int s = 128; s > 0; s >>= 1) { if (t < s) red[t] = fmaxf(red[t], red[t + s]); __syncthreads(); }
  mx = red[0]; __syncthreads();

  float sum = 0.0f;
  #pragma unroll
  for (int i = 0; i < 8; ++i) { vals[i] = __expf(vals[i] - mx); sum += vals[i]; }
  red[t] = sum; __syncthreads();
  for (int s = 128; s > 0; s >>= 1) { if (t < s) red[t] += red[t + s]; __syncthreads(); }
  float inv = 1.0f / red[0];

  #pragma unroll
  for (int i = 0; i < 8; ++i)
    pAlphaT[((size_t)b * P_ + p) * L_ + (t + i * 256)] = vals[i] * inv;
}

// Inclusive prefix sum over L per (b,p) row of pAlphaT.
__global__ __launch_bounds__(256)
void cumsum_L(const float* __restrict__ pAlphaT, float* __restrict__ CS)
{
  const int row = blockIdx.x;                         // b*P + p
  const float* src = pAlphaT + (size_t)row * L_;
  float* dst = CS + (size_t)row * L_;
  const int t = threadIdx.x;
  __shared__ float ts[256];

  float loc[8]; float s = 0.0f;
  const int base = t * 8;
  #pragma unroll
  for (int i = 0; i < 8; ++i) { s += src[base + i]; loc[i] = s; }
  ts[t] = s; __syncthreads();
  for (int off = 1; off < 256; off <<= 1) {
    float v = ts[t];
    float a = (t >= off) ? ts[t - off] : 0.0f;
    __syncthreads();
    ts[t] = v + a;
    __syncthreads();
  }
  float excl = (t > 0) ? ts[t - 1] : 0.0f;
  #pragma unroll
  for (int i = 0; i < 8; ++i) dst[base + i] = loc[i] + excl;
}

// Relative-position modulation via 15 prefix-sum bands + softmax over P.
// scores (d_out region) is read (raw q.k/sqrt), modulated, written back;
// alpha written for the z = alpha @ v GEMM.
__global__ __launch_bounds__(64)
void relmod_softmax_P(const float* __restrict__ CS,    // [B,P,L]
                      const float* __restrict__ embK,  // [15,H]
                      const float* __restrict__ embB,  // [15,H]
                      float* __restrict__ scores,      // [B,H,L,P], in/out
                      float* __restrict__ alpha)       // [B,H,L,P]
{
  const int b = blockIdx.x / L_;
  const int l = blockIdx.x % L_;
  const int p = threadIdx.x;                 // 0..63

  __shared__ float eK[R_ * H_], eB[R_ * H_];
  __shared__ float sm[H_][P_];
  __shared__ float hmax[H_], hsum[H_];
  for (int i = p; i < R_ * H_; i += 64) { eK[i] = embK[i]; eB[i] = embB[i]; }
  __syncthreads();

  const float* cs = CS + ((size_t)b * P_ + p) * L_;

  float BS[R_];
  #pragma unroll
  for (int rr = 0; rr < R_; ++rr) {
    int dlo = c_dlo[rr], dhi = c_dhi[rr];
    float sL = 0.0f;
    int hiI = l - dlo;
    if (hiI >= 0) {
      int loI = l - dhi; if (loI < 0) loI = 0;
      sL = cs[hiI] - (loI > 0 ? cs[loI - 1] : 0.0f);
    }
    float sR = 0.0f;
    if (dlo > 0) {
      int loI = l + dlo;
      if (loI < L_) {
        int hiI2 = l + dhi; if (hiI2 > L_ - 1) hiI2 = L_ - 1;
        sR = cs[hiI2] - cs[loI - 1];
      }
    }
    BS[rr] = sL + sR;
  }

  float Kt[H_], Bt[H_];
  #pragma unroll
  for (int h = 0; h < H_; ++h) { Kt[h] = 0.0f; Bt[h] = 0.0f; }
  #pragma unroll
  for (int rr = 0; rr < R_; ++rr)
    #pragma unroll
    for (int h = 0; h < H_; ++h) {
      Kt[h] = fmaf(eK[rr * H_ + h], BS[rr], Kt[h]);
      Bt[h] = fmaf(eB[rr * H_ + h], BS[rr], Bt[h]);
    }

  float sv[H_];
  #pragma unroll
  for (int h = 0; h < H_; ++h) {
    size_t idx = (((size_t)b * H_ + h) * L_ + l) * P_ + p;
    float s = scores[idx];
    s = s * Kt[h] + Bt[h];
    scores[idx] = s;         // reference returns modulated (pre-softmax) scores
    sv[h] = s;
    sm[h][p] = s;
  }
  __syncthreads();
  if (p < H_) { float m = -3.4e38f; for (int j = 0; j < P_; ++j) m = fmaxf(m, sm[p][j]); hmax[p] = m; }
  __syncthreads();
  float ev[H_];
  #pragma unroll
  for (int h = 0; h < H_; ++h) { ev[h] = __expf(sv[h] - hmax[h]); sm[h][p] = ev[h]; }
  __syncthreads();
  if (p < H_) { float s = 0.0f; for (int j = 0; j < P_; ++j) s += sm[p][j]; hsum[p] = s; }
  __syncthreads();
  #pragma unroll
  for (int h = 0; h < H_; ++h)
    alpha[(((size_t)b * H_ + h) * L_ + l) * P_ + p] = ev[h] / hsum[h];
}

extern "C" void kernel_launch(void* const* d_in, const int* in_sizes, int n_in,
                              void* d_out, int out_size, void* d_ws, size_t ws_size,
                              hipStream_t stream)
{
  const float* qx      = (const float*)d_in[0];
  const float* kx      = (const float*)d_in[1];
  const float* vx      = (const float*)d_in[2];
  const int*   maskPAD = (const int*)  d_in[3];
  const float* Wp = (const float*)d_in[4];
  const float* bp = (const float*)d_in[5];
  const float* Wq = (const float*)d_in[6];
  const float* bq = (const float*)d_in[7];
  const float* Wk = (const float*)d_in[8];
  const float* bk = (const float*)d_in[9];
  const float* Wv = (const float*)d_in[10];
  const float* bv = (const float*)d_in[11];
  const float* Wo = (const float*)d_in[12];
  const float* bo = (const float*)d_in[13];
  const float* embK = (const float*)d_in[14];
  const float* embB = (const float*)d_in[15];

  float* out    = (float*)d_out;
  float* z_out  = out;                                   // [B,L,C]
  float* pscore = out + (size_t)B_ * L_ * C_;            // [B,L,P]
  float* scores = pscore + (size_t)B_ * L_ * P_;         // [B,H,L,P]

  float* w = (float*)d_ws;
  size_t o = 0;
  float* pAlphaT = w + o; o += (size_t)B_ * P_ * L_;
  float* CS      = w + o; o += (size_t)B_ * P_ * L_;
  float* qbuf    = w + o; o += (size_t)B_ * L_ * HD_;
  float* kxc     = w + o; o += (size_t)B_ * P_ * C_;
  float* vxc     = w + o; o += (size_t)B_ * P_ * C_;
  float* kk      = w + o; o += (size_t)B_ * P_ * HD_;
  float* vv      = w + o; o += (size_t)B_ * P_ * HD_;
  float* alpha   = w + o; o += (size_t)B_ * H_ * L_ * P_;
  float* zpre    = qbuf;  // q is dead after the scores GEMM; reuse buffer

  const dim3 blk(64);

  // 1) pScore_raw = vx @ Wp + bp                    [16384,512]x[512,64]
  gemm_bf16_wmma<<<dim3(P_ / TN, (B_ * L_) / TM, 1), blk, 0, stream>>>(
      vx, C_, 0, 0, Wp, P_, 0, 0, 0,
      pscore, P_, 0, 0, bp, 1.0f, B_ * L_, P_, C_, 1);

  // 2) q = qx @ Wq + bq                             [16384,512]x[512,512]
  gemm_bf16_wmma<<<dim3(HD_ / TN, (B_ * L_) / TM, 1), blk, 0, stream>>>(
      qx, C_, 0, 0, Wq, HD_, 0, 0, 0,
      qbuf, HD_, 0, 0, bq, 1.0f, B_ * L_, HD_, C_, 1);

  // 3) mask + softmax over L -> pAlphaT [B,P,L]
  mask_softmax_L<<<dim3(B_ * P_), dim3(256), 0, stream>>>(pscore, maskPAD, pAlphaT);

  // 4) cumsum over L
  cumsum_L<<<dim3(B_ * P_), dim3(256), 0, stream>>>(pAlphaT, CS);

  // 5) kxc = pAlpha^T @ kx  (per batch)             [64,2048]x[2048,512]
  gemm_bf16_wmma<<<dim3(C_ / TN, P_ / TM, B_), blk, 0, stream>>>(
      pAlphaT, L_, (long long)P_ * L_, 0, kx, C_, (long long)L_ * C_, 0, 0,
      kxc, C_, (long long)P_ * C_, 0, nullptr, 1.0f, P_, C_, L_, 1);

  // 6) vxc = pAlpha^T @ vx
  gemm_bf16_wmma<<<dim3(C_ / TN, P_ / TM, B_), blk, 0, stream>>>(
      pAlphaT, L_, (long long)P_ * L_, 0, vx, C_, (long long)L_ * C_, 0, 0,
      vxc, C_, (long long)P_ * C_, 0, nullptr, 1.0f, P_, C_, L_, 1);

  // 7) k = kxc @ Wk + bk (batch flattened into M)   [512,512]x[512,512]
  gemm_bf16_wmma<<<dim3(HD_ / TN, (B_ * P_) / TM, 1), blk, 0, stream>>>(
      kxc, C_, 0, 0, Wk, HD_, 0, 0, 0,
      kk, HD_, 0, 0, bk, 1.0f, B_ * P_, HD_, C_, 1);

  // 8) v = vxc @ Wv + bv
  gemm_bf16_wmma<<<dim3(HD_ / TN, (B_ * P_) / TM, 1), blk, 0, stream>>>(
      vxc, C_, 0, 0, Wv, HD_, 0, 0, 0,
      vv, HD_, 0, 0, bv, 1.0f, B_ * P_, HD_, C_, 1);

  // 9) scores_raw = q . k^T / sqrt(DK)  per (b,h)   [2048,64]x[64,64], z=64
  gemm_bf16_wmma<<<dim3(P_ / TN, L_ / TM, B_ * H_), blk, 0, stream>>>(
      qbuf, HD_, (long long)L_ * HD_, DK_,
      kk,   HD_, (long long)P_ * HD_, DK_, 1 /*transB*/,
      scores, P_, (long long)H_ * L_ * P_, (long long)L_ * P_,
      nullptr, 0.125f, L_, P_, DK_, H_);

  // 10) relative-position modulation + softmax over P
  relmod_softmax_P<<<dim3(B_ * L_), dim3(64), 0, stream>>>(CS, embK, embB, scores, alpha);

  // 11) z_pre = alpha @ v  per (b,h)                [2048,64]x[64,64], z=64
  gemm_bf16_wmma<<<dim3(DK_ / TN, L_ / TM, B_ * H_), blk, 0, stream>>>(
      alpha, P_, (long long)H_ * L_ * P_, (long long)L_ * P_,
      vv,    HD_, (long long)P_ * HD_, DK_, 0,
      zpre, HD_, (long long)L_ * HD_, DK_,
      nullptr, 1.0f, L_, DK_, P_, H_);

  // 12) z = z_pre @ Wo + bo                         [16384,512]x[512,512]
  gemm_bf16_wmma<<<dim3(C_ / TN, (B_ * L_) / TM, 1), blk, 0, stream>>>(
      zpre, HD_, 0, 0, Wo, C_, 0, 0, 0,
      z_out, C_, 0, 0, bo, 1.0f, B_ * L_, C_, HD_, 1);
}
